// KNRM_5875515261166
// MI455X (gfx1250) — matrix operand order repaired and checked by reading, not compiled, not run on hardware
//
#include <hip/hip_runtime.h>

typedef _Float16 v16h __attribute__((ext_vector_type(16)));
typedef _Float16 v8h  __attribute__((ext_vector_type(8)));
typedef _Float16 v4h  __attribute__((ext_vector_type(4)));
typedef float    v8f  __attribute__((ext_vector_type(8)));
typedef float    v4f  __attribute__((ext_vector_type(4)));

#define B_   256
#define Q_   20
#define D_   512
#define E_   300
#define E4_  75    // E_ in float4 units
#define EP4_ 80    // padded K (320) in 4-element units
#define ST_  328   // K stride in halves: 656B rows -> 16B aligned, conflict-free
#define CH_  64    // docs per chunk
#define NCH_ 8

#define QE_HALFS (32 * ST_)
#define DE_HALFS (CH_ * ST_)
#define PKQP_F   (4 * 32 * 12)   // [doc-tile group][q row][kernel]

static constexpr unsigned SMEM_BYTES =
    (QE_HALFS + DE_HALFS) * 2 + (PKQP_F + 32 + CH_) * 4 + (32 + CH_) * 4;

// Gather one embedding row (float4-wide), L2-normalize, store as f16 into LDS
// tile row `dst` (stride ST_ halves). Covers K 0..319 incl. zero padding.
__device__ __forceinline__ void gather_norm_row(const float* __restrict__ emb,
                                                int tok, _Float16* dst, int lane) {
  const v4f* rowp = (const v4f*)(emb + (size_t)tok * E_);
  v4f vals[3];
  float ss = 0.f;
  #pragma unroll
  for (int i = 0; i < 3; ++i) {
    int idx = lane + i * 32;
    v4f v = {};
    if (idx < E4_) v = rowp[idx];          // global_load_b128, coalesced
    vals[i] = v;
    ss += v.x * v.x + v.y * v.y + v.z * v.z + v.w * v.w;
  }
  #pragma unroll
  for (int off = 16; off >= 1; off >>= 1) ss += __shfl_xor(ss, off, 32);
  float inv = 1.f / fmaxf(sqrtf(ss), 1e-13f);
  #pragma unroll
  for (int i = 0; i < 3; ++i) {
    int idx = lane + i * 32;
    if (idx < EP4_) {                      // idx 75..79 store zeros = K padding
      v4f v = vals[i] * inv;
      *(v4h*)(dst + idx * 4) = __builtin_convertvector(v, v4h);  // ds_store_b64
    }
  }
}

__global__ __launch_bounds__(256) void knrm_fused(
    const int* __restrict__ qtok, const int* __restrict__ dtok,
    const float* __restrict__ emb, const float* __restrict__ w,
    const float* __restrict__ bias, float* __restrict__ out)
{
  extern __shared__ __attribute__((aligned(16))) char smem[];
  _Float16* qE     = (_Float16*)smem;          // 32 x ST_
  _Float16* dE     = qE + QE_HALFS;            // 64 x ST_
  float*    pkqp   = (float*)(dE + DE_HALFS);  // 4 x 32 x 12 partial sums
  float*    qmaskL = pkqp + PKQP_F;            // 32
  float*    dmaskL = qmaskL + 32;              // 64
  int*      qtokL  = (int*)(dmaskL + CH_);     // 32
  int*      dtokL  = qtokL + 32;               // 64

  const int b    = blockIdx.x;
  const int tid  = threadIdx.x;
  const int lane = tid & 31;
  const int wave = tid >> 5;
  const int half = lane >> 4;
  const int l15  = lane & 15;

  // ---------------- query phase: gather + normalize to f16 LDS ----------------
  for (int r = wave * 4; r < wave * 4 + 4; ++r) {
    if (r < Q_) {
      int tok = qtok[b * Q_ + r];
      gather_norm_row(emb, tok, qE + (size_t)r * ST_, lane);
      if (lane == 0) { qmaskL[r] = (tok > 0) ? 1.f : 0.f; qtokL[r] = tok; }
    } else {  // padding rows 20..31: zero A rows
      #pragma unroll
      for (int i = 0; i < 3; ++i) {
        int idx = lane + i * 32;
        if (idx < EP4_) *(v4h*)(qE + (size_t)r * ST_ + idx * 4) = (v4h){};
      }
      if (lane == 0) { qmaskL[r] = 0.f; qtokL[r] = -1; }
    }
  }
  __syncthreads();

  const int qtile = wave & 1;   // q rows [qtile*16, qtile*16+16)
  const int dtile = wave >> 1;  // doc tile 0..3 within chunk

  float qmv[8]; int qtv[8];
  #pragma unroll
  for (int v = 0; v < 8; ++v) {
    int row = qtile * 16 + v + 8 * half;  // C-layout: vgpr v -> M = v + 8*(lane/16)
    qmv[v] = qmaskL[row];
    qtv[v] = qtokL[row];
  }

  float acc[8][10];  // Gaussian kernels k=1..10 (sigma=0.1 -> 1/(2s^2)=50)
  float acc0[8];     // exact-match kernel (sigma=1e-4) via token equality
  #pragma unroll
  for (int v = 0; v < 8; ++v) {
    acc0[v] = 0.f;
    #pragma unroll
    for (int k = 0; k < 10; ++k) acc[v][k] = 0.f;
  }

  for (int chunk = 0; chunk < NCH_; ++chunk) {
    __syncthreads();  // previous chunk's dE readers done
    // ---- gather + normalize 64 docs (8 per wave) ----
    for (int j = 0; j < 8; ++j) {
      int dd  = wave * 8 + j;
      int tok = dtok[b * D_ + chunk * CH_ + dd];
      gather_norm_row(emb, tok, dE + (size_t)dd * ST_, lane);
      if (lane == 0) { dmaskL[dd] = (tok > 0) ? 1.f : 0.f; dtokL[dd] = tok; }
    }
    __syncthreads();

    // ---- 16x16 cosine tile via WMMA f16, K = 320 in 10 steps ----
    v8f c = {};
    // A frag (16-bit A layout): lane l15=M, VGPR0-3 = K kb..kb+7, VGPR4-7 = K kb+16..kb+23, kb = 32s + 8*half
    const _Float16* arow = qE + (size_t)(qtile * 16 + l15) * ST_ + 8 * half;
    // B frag (lane = column N, same K layout as A: K kb..kb+7, kb+16..kb+23, kb = 32s + 8*half)
    const _Float16* brow = dE + (size_t)(dtile * 16 + l15) * ST_ + 8 * half;
    #pragma unroll
    for (int s = 0; s < 10; ++s) {
      v8h a0 = *(const v8h*)(arow + s * 32);
      v8h a1 = *(const v8h*)(arow + s * 32 + 16);
      v8h b0 = *(const v8h*)(brow + s * 32);
      v8h b1 = *(const v8h*)(brow + s * 32 + 16);
      v16h A  = __builtin_shufflevector(a0, a1, 0,1,2,3,4,5,6,7,8,9,10,11,12,13,14,15);
      v16h Bm = __builtin_shufflevector(b0, b1, 0,1,2,3,4,5,6,7,8,9,10,11,12,13,14,15);
      c = __builtin_amdgcn_wmma_f32_16x16x32_f16(false, A, false, Bm, (short)0, c,
                                                 false, false);
    }

    // ---- Gaussian kernel pooling, accumulate over this doc tile ----
    float dm = dmaskL[dtile * 16 + l15];
    int   dt = dtokL[dtile * 16 + l15];
    const float mu[10] = {0.9f, 0.7f, 0.5f, 0.3f, 0.1f,
                          -0.1f, -0.3f, -0.5f, -0.7f, -0.9f};
    #pragma unroll
    for (int v = 0; v < 8; ++v) {
      float m  = qmv[v] * dm;
      float cv = c[v];
      acc0[v] += (qtv[v] == dt) ? m : 0.f;   // sigma=1e-4 kernel == exact token match
      #pragma unroll
      for (int k = 0; k < 10; ++k) {
        float d1 = cv - mu[k];
        acc[v][k] += m * __expf(-50.f * d1 * d1);
      }
    }
  }

  // ---- deterministic reduction: 16 lanes share a row; write per-dtile partials ----
  #pragma unroll
  for (int v = 0; v < 8; ++v) {
    int row = qtile * 16 + v + 8 * half;
    float t = acc0[v];
    t += __shfl_xor(t, 1, 16); t += __shfl_xor(t, 2, 16);
    t += __shfl_xor(t, 4, 16); t += __shfl_xor(t, 8, 16);
    if (l15 == 0) pkqp[(dtile * 32 + row) * 12 + 0] = t;
    #pragma unroll
    for (int k = 0; k < 10; ++k) {
      float u = acc[v][k];
      u += __shfl_xor(u, 1, 16); u += __shfl_xor(u, 2, 16);
      u += __shfl_xor(u, 4, 16); u += __shfl_xor(u, 8, 16);
      if (l15 == 0) pkqp[(dtile * 32 + row) * 12 + (k + 1)] = u;
    }
  }
  __syncthreads();

  // ---- log pooling over Q + dense layer ----
  if (tid == 0) {
    float res = bias[0];
    for (int k = 0; k < 11; ++k) {
      float s = 0.f;
      for (int q = 0; q < Q_; ++q) {
        if (qmaskL[q] > 0.f) {
          float val = pkqp[(0 * 32 + q) * 12 + k] + pkqp[(1 * 32 + q) * 12 + k] +
                      pkqp[(2 * 32 + q) * 12 + k] + pkqp[(3 * 32 + q) * 12 + k];
          s += __logf(fmaxf(val, 1e-10f)) * 0.01f;
        }
      }
      res += s * w[k];
    }
    out[b] = res;
  }
}

extern "C" void kernel_launch(void* const* d_in, const int* in_sizes, int n_in,
                              void* d_out, int out_size, void* d_ws, size_t ws_size,
                              hipStream_t stream) {
  (void)in_sizes; (void)n_in; (void)d_ws; (void)ws_size; (void)out_size;
  const int*   qt   = (const int*)d_in[0];
  const int*   dt   = (const int*)d_in[1];
  const float* emb  = (const float*)d_in[2];
  const float* w    = (const float*)d_in[3];
  const float* bias = (const float*)d_in[4];
  float*       out  = (float*)d_out;

  hipFuncSetAttribute((const void*)knrm_fused,
                      hipFuncAttributeMaxDynamicSharedMemorySize,
                      (int)SMEM_BYTES);
  knrm_fused<<<dim3(B_), dim3(256), SMEM_BYTES, stream>>>(qt, dt, emb, w, bias, out);
}
